// MultiHeadAttention_996432413295
// MI455X (gfx1250) — matrix operand order repaired and checked by reading, not compiled, not run on hardware
//
#include <hip/hip_runtime.h>
#include <hip/hip_bf16.h>

// ---------------------------------------------------------------------------
// MHA forward on gfx1250 (MI455X): bf16 WMMA everywhere, f32 accumulate.
// ---------------------------------------------------------------------------

typedef unsigned short ushort_t;
typedef __attribute__((ext_vector_type(16))) __bf16 bf16x16;
typedef __attribute__((ext_vector_type(8)))  float  f32x8;

#define EMBED 1024
#define HEADS 16
#define HDIM  64
#define BATCH 2
#define SEQ   2048
#define MROWS (BATCH * SEQ) /* 4096 */

union FragBF {
  bf16x16 v;
  uint4   q[2];
};

__device__ __forceinline__ ushort_t f2bf(float f) {
  unsigned u = __float_as_uint(f);
  u += 0x7FFFu + ((u >> 16) & 1u); // round-to-nearest-even
  return (ushort_t)(u >> 16);
}

__device__ __forceinline__ f32x8 wmma_bf16(const FragBF& a, const FragBF& b, f32x8 c) {
  return __builtin_amdgcn_wmma_f32_16x16x32_bf16(false, a.v, false, b.v,
                                                 (short)0, c, false, false);
}

__device__ __forceinline__ float rmax16(float v) {
#pragma unroll
  for (int m = 1; m < 16; m <<= 1) v = fmaxf(v, __shfl_xor(v, m, 16));
  return v;
}
__device__ __forceinline__ float rsum16(float v) {
#pragma unroll
  for (int m = 1; m < 16; m <<= 1) v += __shfl_xor(v, m, 16);
  return v;
}

// ---------------------------------------------------------------------------
// f32 -> bf16 conversion (vectorized x4)
// ---------------------------------------------------------------------------
__global__ __launch_bounds__(256) void mha_cvt_bf16(const float* __restrict__ src,
                                                    ushort_t* __restrict__ dst, int n) {
  int i = (blockIdx.x * 256 + threadIdx.x) * 4;
  if (i + 3 < n) {
    float4 f = *(const float4*)(src + i);
    uint2 o;
    o.x = (unsigned)f2bf(f.x) | ((unsigned)f2bf(f.y) << 16);
    o.y = (unsigned)f2bf(f.z) | ((unsigned)f2bf(f.w) << 16);
    *(uint2*)(dst + i) = o;
  }
}

// ---------------------------------------------------------------------------
// GEMM: out[m,n] = sum_k A[m,k] * W[n,k] + bias[n]     (x @ W^T + b)
// A: MROWS x 1024 bf16 row-major, W: 1024 x 1024 bf16 row-major.
// Wave computes 32x64 (2x4 16x16 D-tiles); block = 8 waves = 128M x 128N.
// mode 0: bf16 out [B,H,S,64]   (Q, K)
// mode 1: bf16 out [B,H,64,S]   (V transposed for PV B-fragments)
// mode 2: f32  out [MROWS,1024] (final projection -> d_out)
// ---------------------------------------------------------------------------
__global__ __launch_bounds__(256)
void mha_gemm_bf16(const ushort_t* __restrict__ A, const ushort_t* __restrict__ W,
                   const float* __restrict__ bias, ushort_t* __restrict__ outb,
                   float* __restrict__ outf, int mode) {
  const int K    = EMBED;
  const int lane = threadIdx.x & 31;
  const int wave = threadIdx.x >> 5;
  const int l16  = lane & 15;
  const int half = lane >> 4;

  const int m0 = blockIdx.y * 128 + (wave & 3) * 32;
  const int n0 = blockIdx.x * 128 + (wave >> 2) * 64;

  // A-fragment rows (lane m = l16, 16 contiguous K at half*16)
  const ushort_t* aRow0 = A + (size_t)(m0 + l16) * K + half * 16;
  const ushort_t* aRow1 = aRow0 + (size_t)16 * K;
  // B-fragment rows (lane n = l16, 16 contiguous K at half*16, same k order as A)
  const ushort_t* wRow[4];
#pragma unroll
  for (int j = 0; j < 4; ++j)
    wRow[j] = W + (size_t)(n0 + 16 * j + l16) * K + half * 16;

  f32x8 acc[2][4];
#pragma unroll
  for (int i = 0; i < 2; ++i)
#pragma unroll
    for (int j = 0; j < 4; ++j)
      acc[i][j] = (f32x8){0.f, 0.f, 0.f, 0.f, 0.f, 0.f, 0.f, 0.f};

  for (int kc = 0; kc < K; kc += 32) {
    FragBF a0, a1;
    a0.q[0] = *(const uint4*)(aRow0 + kc);
    a0.q[1] = *(const uint4*)(aRow0 + kc + 8);
    a1.q[0] = *(const uint4*)(aRow1 + kc);
    a1.q[1] = *(const uint4*)(aRow1 + kc + 8);
#pragma unroll
    for (int j = 0; j < 4; ++j) {
      FragBF bw;
      bw.q[0] = *(const uint4*)(wRow[j] + kc);
      bw.q[1] = *(const uint4*)(wRow[j] + kc + 8);
      acc[0][j] = wmma_bf16(a0, bw, acc[0][j]);
      acc[1][j] = wmma_bf16(a1, bw, acc[1][j]);
    }
  }

  // Epilogue. D layout: lane column n = l16; VGPR r = row half*8+r.
#pragma unroll
  for (int j = 0; j < 4; ++j) {
    const int n  = n0 + 16 * j + l16;
    const float bn = bias[n];
    const int h  = n >> 6;
    const int dd = n & 63;
#pragma unroll
    for (int i = 0; i < 2; ++i) {
#pragma unroll
      for (int r = 0; r < 8; ++r) {
        const int m = m0 + i * 16 + half * 8 + r;
        const float val = acc[i][j][r] + bn;
        if (mode == 2) {
          outf[(size_t)m * EMBED + n] = val;
        } else {
          const int b = m >> 11;       // m = b*SEQ + s
          const int s = m & (SEQ - 1);
          if (mode == 0)
            outb[(((size_t)(b * HEADS + h)) * SEQ + s) * HDIM + dd] = f2bf(val);
          else // mode 1: V transposed [B,H,64,S]
            outb[(((size_t)(b * HEADS + h)) * HDIM + dd) * SEQ + s] = f2bf(val);
        }
      }
    }
  }
}

// ---------------------------------------------------------------------------
// Flash attention: one wave = one 16-query tile of one (b,h).
// Streams 32-key blocks: 4 WMMA (QK^T over d=64) + online softmax
// + P via LDS (D-layout -> A-layout) + 4 WMMA (PV into 16x64 O accumulator).
// ---------------------------------------------------------------------------
__global__ __launch_bounds__(128)
void mha_flash_attn(const ushort_t* __restrict__ Q, const ushort_t* __restrict__ Kb,
                    const ushort_t* __restrict__ Vt, const int* __restrict__ mask,
                    ushort_t* __restrict__ AO) {
  __shared__ __align__(16) ushort_t pbuf[4][16 * 32]; // 1KB per wave

  const int lane = threadIdx.x & 31;
  const int wave = threadIdx.x >> 5;
  const int l16  = lane & 15;
  const int half = lane >> 4;

  const int t  = blockIdx.x * 4 + wave;  // 0..4095
  const int bh = t >> 7;                 // (b*16+h)
  const int q0 = (t & 127) * 16;
  const int b  = bh >> 4;
  const int h  = bh & 15;

  // Q A-fragments for dims 0..31 and 32..63
  const ushort_t* qbase = Q + ((size_t)bh * SEQ + q0 + l16) * HDIM;
  FragBF qa[2];
#pragma unroll
  for (int j = 0; j < 2; ++j) {
    qa[j].q[0] = *(const uint4*)(qbase + j * 32 + half * 16);
    qa[j].q[1] = *(const uint4*)(qbase + j * 32 + half * 16 + 8);
  }

  const ushort_t* kbh = Kb + (size_t)bh * SEQ * HDIM;
  const ushort_t* vbh = Vt + (size_t)bh * HDIM * SEQ;
  const int* mrowp    = mask + b * SEQ;
  ushort_t* pb        = pbuf[wave];

  float mrow[8], lrow[8];
  f32x8 o[4];
#pragma unroll
  for (int r = 0; r < 8; ++r) { mrow[r] = -INFINITY; lrow[r] = 0.f; }
#pragma unroll
  for (int d = 0; d < 4; ++d) o[d] = (f32x8){0.f, 0.f, 0.f, 0.f, 0.f, 0.f, 0.f, 0.f};

  for (int kb0 = 0; kb0 < SEQ; kb0 += 32) {
    // ---- scores: two 16x16 tiles (keys kb0..+15 and kb0+16..+31) ----
    f32x8 s0 = (f32x8){0.f, 0.f, 0.f, 0.f, 0.f, 0.f, 0.f, 0.f};
    f32x8 s1 = s0;
    {
      const ushort_t* kp0 = kbh + (size_t)(kb0 + l16) * HDIM + half * 16;
      const ushort_t* kp1 = kp0 + 16 * HDIM;
      FragBF kf;
      kf.q[0] = *(const uint4*)(kp0);      kf.q[1] = *(const uint4*)(kp0 + 8);
      s0 = wmma_bf16(qa[0], kf, s0);
      kf.q[0] = *(const uint4*)(kp0 + 32); kf.q[1] = *(const uint4*)(kp0 + 40);
      s0 = wmma_bf16(qa[1], kf, s0);
      kf.q[0] = *(const uint4*)(kp1);      kf.q[1] = *(const uint4*)(kp1 + 8);
      s1 = wmma_bf16(qa[0], kf, s1);
      kf.q[0] = *(const uint4*)(kp1 + 32); kf.q[1] = *(const uint4*)(kp1 + 40);
      s1 = wmma_bf16(qa[1], kf, s1);
    }

    // ---- scale + key mask (column = lane) ----
    const int mk0 = mrowp[kb0 + l16];
    const int mk1 = mrowp[kb0 + 16 + l16];
    const float SC = 0.125f; // 1/sqrt(64)
#pragma unroll
    for (int r = 0; r < 8; ++r) {
      s0[r] = mk0 ? s0[r] * SC : -1e9f;
      s1[r] = mk1 ? s1[r] * SC : -1e9f;
    }

    // ---- online softmax (row = half*8+r spread over 16 lanes of this half) ----
    float p0[8], p1[8], alpha[8];
#pragma unroll
    for (int r = 0; r < 8; ++r) {
      float bm = rmax16(fmaxf(s0[r], s1[r]));
      float mn = fmaxf(mrow[r], bm);
      alpha[r] = __expf(mrow[r] - mn);
      mrow[r]  = mn;
      p0[r] = __expf(s0[r] - mn);
      p1[r] = __expf(s1[r] - mn);
      lrow[r] = lrow[r] * alpha[r] + rsum16(p0[r] + p1[r]);
    }
#pragma unroll
    for (int d = 0; d < 4; ++d)
#pragma unroll
      for (int r = 0; r < 8; ++r) o[d][r] *= alpha[r];

    // ---- P: D-layout regs -> LDS row-major 16x32 -> A-fragment ----
#pragma unroll
    for (int r = 0; r < 8; ++r) {
      pb[(half * 8 + r) * 32 + l16]      = f2bf(p0[r]);
      pb[(half * 8 + r) * 32 + 16 + l16] = f2bf(p1[r]);
    }
    FragBF pf; // same-wave LDS RAW; compiler inserts s_wait_dscnt
    pf.q[0] = *(const uint4*)&pb[l16 * 32 + half * 16];
    pf.q[1] = *(const uint4*)&pb[l16 * 32 + half * 16 + 8];

    // ---- O += P @ V (4 d-tiles; Vt makes k-contiguous 32B B-fragments) ----
    const ushort_t* vcol = vbh + (size_t)l16 * SEQ + kb0 + half * 16;
#pragma unroll
    for (int d = 0; d < 4; ++d) {
      const ushort_t* vp = vcol + (size_t)(d * 16) * SEQ;
      FragBF vf;
      vf.q[0] = *(const uint4*)(vp);
      vf.q[1] = *(const uint4*)(vp + 8);
      o[d] = wmma_bf16(pf, vf, o[d]);
    }
  }

  // ---- epilogue: O/l -> bf16 attn-out [B,S,1024] for the final projection ----
#pragma unroll
  for (int d = 0; d < 4; ++d) {
#pragma unroll
    for (int r = 0; r < 8; ++r) {
      const int s = q0 + half * 8 + r;
      const float val = o[d][r] / lrow[r];
      AO[((size_t)(b * SEQ + s)) * EMBED + h * HDIM + d * 16 + l16] = f2bf(val);
    }
  }
}

// ---------------------------------------------------------------------------
// Host-side orchestration
// ---------------------------------------------------------------------------
extern "C" void kernel_launch(void* const* d_in, const int* in_sizes, int n_in,
                              void* d_out, int out_size, void* d_ws, size_t ws_size,
                              hipStream_t stream) {
  const float* x    = (const float*)d_in[0];
  const int*   mask = (const int*)d_in[1];
  const float* Wq   = (const float*)d_in[2];
  const float* bq   = (const float*)d_in[3];
  const float* Wk   = (const float*)d_in[4];
  const float* bk   = (const float*)d_in[5];
  const float* Wv   = (const float*)d_in[6];
  const float* bv   = (const float*)d_in[7];
  const float* Wo   = (const float*)d_in[8];
  const float* bo   = (const float*)d_in[9];
  float* out = (float*)d_out;

  const size_t nX = (size_t)MROWS * EMBED;      // 4,194,304
  const size_t nW = (size_t)EMBED * EMBED;      // 1,048,576
  char* ws = (char*)d_ws;
  size_t off = 0;
  ushort_t* XB  = (ushort_t*)(ws + off); off += nX * 2;
  ushort_t* WQB = (ushort_t*)(ws + off); off += nW * 2;
  ushort_t* WKB = (ushort_t*)(ws + off); off += nW * 2;
  ushort_t* WVB = (ushort_t*)(ws + off); off += nW * 2;
  ushort_t* WOB = (ushort_t*)(ws + off); off += nW * 2;
  ushort_t* QB  = (ushort_t*)(ws + off); off += nX * 2;
  ushort_t* KB  = (ushort_t*)(ws + off); off += nX * 2;
  ushort_t* VTB = (ushort_t*)(ws + off); off += nX * 2;
  ushort_t* AOB = (ushort_t*)(ws + off); off += nX * 2;
  if (ws_size < off) return; // needs ~48 MB scratch

  // 1) convert inputs to bf16
  mha_cvt_bf16<<<(int)(nX / 1024), 256, 0, stream>>>(x,  XB,  (int)nX);
  mha_cvt_bf16<<<(int)(nW / 1024), 256, 0, stream>>>(Wq, WQB, (int)nW);
  mha_cvt_bf16<<<(int)(nW / 1024), 256, 0, stream>>>(Wk, WKB, (int)nW);
  mha_cvt_bf16<<<(int)(nW / 1024), 256, 0, stream>>>(Wv, WVB, (int)nW);
  mha_cvt_bf16<<<(int)(nW / 1024), 256, 0, stream>>>(Wo, WOB, (int)nW);

  // 2) projections (WMMA GEMMs)
  dim3 gGemm(EMBED / 128, MROWS / 128); // (8, 32)
  mha_gemm_bf16<<<gGemm, 256, 0, stream>>>(XB, WQB, bq, QB,  nullptr, 0);
  mha_gemm_bf16<<<gGemm, 256, 0, stream>>>(XB, WKB, bk, KB,  nullptr, 0);
  mha_gemm_bf16<<<gGemm, 256, 0, stream>>>(XB, WVB, bv, VTB, nullptr, 1);

  // 3) flash attention: 4096 16-query tiles, 4 waves/block
  mha_flash_attn<<<(BATCH * HEADS * (SEQ / 16)) / 4, 128, 0, stream>>>(
      QB, KB, VTB, mask, AOB);

  // 4) output projection -> f32 d_out
  mha_gemm_bf16<<<gGemm, 256, 0, stream>>>(AOB, WOB, bo, nullptr, out, 2);
}
